// ETNGenerator_44495861186791
// MI455X (gfx1250) — matrix operand order repaired and checked by reading, not compiled, not run on hardware
//
#include <hip/hip_runtime.h>

typedef __attribute__((ext_vector_type(16))) __bf16 v16bf;
typedef __attribute__((ext_vector_type(8)))  float  v8f;

#define NT 256
#define NWAVE 8

// ---------------------------------------------------------------------------
// Weight re-tiling for WMMA A-fragments (this code's contiguous k order, B uses the same):
//   lanes 0-15 hold K {0..15}, lanes 16-31 hold K {16..31};
//   v16bf element e -> K = 16*(lane>=16) + e
// Tile storage: dst[((rt*nkt)+kt)*512 + lane*16 + e]
// ---------------------------------------------------------------------------
__device__ __forceinline__ void tile_decompose(int li, int& m, int& kl) {
  int lane = li >> 4, e = li & 15;
  m = lane & 15;
  kl = ((lane >= 16) ? 16 : 0) + e;
}

// pred_weights: [B=128][O=2048][K=1280], already row-major per output row.
__global__ void convert_pred(const float* __restrict__ src, __bf16* __restrict__ dst) {
  const size_t total = (size_t)128 * 128 * 40 * 512;
  size_t idx = (size_t)blockIdx.x * NT + threadIdx.x;
  if (idx >= total) return;
  const size_t per_s = (size_t)128 * 40 * 512;
  int s = (int)(idx / per_s);
  int r = (int)(idx % per_s);
  int rt = r / (40 * 512);
  int r2 = r % (40 * 512);
  int kt = r2 / 512;
  int li = r2 % 512;
  int m, kl; tile_decompose(li, m, kl);
  int o = rt * 16 + m;
  int k = kt * 32 + kl;
  dst[idx] = (__bf16)src[((size_t)s * 2048 + o) * 1280 + k];
}

// shared weights stored [K_in][O_out] (x @ W); transpose + zero-pad into tiles
__global__ void convert_wt(const float* __restrict__ W, __bf16* __restrict__ dst,
                           int O, int K, int nrt, int nkt) {
  int total = nrt * nkt * 512;
  int idx = blockIdx.x * NT + threadIdx.x;
  if (idx >= total) return;
  int rt = idx / (nkt * 512);
  int r2 = idx % (nkt * 512);
  int kt = r2 / 512;
  int li = r2 % 512;
  int m, kl; tile_decompose(li, m, kl);
  int o = rt * 16 + m;
  int k = kt * 32 + kl;
  float v = (o < O && k < K) ? W[(size_t)k * O + o] : 0.f;
  dst[idx] = (__bf16)v;
}

// ---------------------------------------------------------------------------
// WMMA matvec: out[0..nrt*16) = act(Wt[nrt*16 x nkt*32] * z + bias)
// B fragment broadcasts z to all 16 N-columns, so every D column equals the
// matvec; lanes 0 and 16 extract rows 0..7 / 8..15 of each row-tile.
// bias lives in LDS (zero-padded), out buffers are padded to 16-row multiples,
// so the epilogue is branch-free and vectorized (float4 ds ops).
// ---------------------------------------------------------------------------
__device__ __forceinline__ float sigmf(float x) { return 1.f / (1.f + __expf(-x)); }

template <bool RELU>
__device__ __forceinline__ void wmma_mv(const __bf16* __restrict__ Wt,
                                        const __bf16* __restrict__ z,
                                        float* __restrict__ out,
                                        const float* __restrict__ bias,  // LDS
                                        int nrt, int nkt,
                                        int lane, int wave) {
  const int half = (lane >= 16) ? 16 : 0;
  for (int rt = wave; rt < nrt; rt += NWAVE) {
    v8f acc = {};
    const __bf16* wp = Wt + (size_t)rt * nkt * 512 + lane * 16;
#pragma unroll 4
    for (int kt = 0; kt < nkt; ++kt) {
      v16bf a = *(const v16bf*)wp;
      v16bf b = *(const v16bf*)(z + kt * 32 + half);          // LDS, 32B aligned
      __builtin_prefetch((const void*)(wp + 512), 0, 1);      // global_prefetch_b8
      acc = __builtin_amdgcn_wmma_f32_16x16x32_bf16(
          false, a, false, b, (short)0, acc, false, false);
      wp += 512;
    }
    if ((lane & 15) == 0) {
      int base = rt * 16 + (half ? 8 : 0);
      float4 b0 = *(const float4*)(bias + base);
      float4 b1 = *(const float4*)(bias + base + 4);
      float4 o0, o1;
      o0.x = acc[0] + b0.x; o0.y = acc[1] + b0.y; o0.z = acc[2] + b0.z; o0.w = acc[3] + b0.w;
      o1.x = acc[4] + b1.x; o1.y = acc[5] + b1.y; o1.z = acc[6] + b1.z; o1.w = acc[7] + b1.w;
      if (RELU) {
        o0.x = fmaxf(o0.x, 0.f); o0.y = fmaxf(o0.y, 0.f);
        o0.z = fmaxf(o0.z, 0.f); o0.w = fmaxf(o0.w, 0.f);
        o1.x = fmaxf(o1.x, 0.f); o1.y = fmaxf(o1.y, 0.f);
        o1.z = fmaxf(o1.z, 0.f); o1.w = fmaxf(o1.w, 0.f);
      }
      *(float4*)(out + base) = o0;
      *(float4*)(out + base + 4) = o1;
    }
  }
}

// LDS bias pool layout (floats); all offsets multiples of 32
#define BOFF_SE1 0
#define BOFF_SE2 512
#define BOFF_OE1 768
#define BOFF_OE2 1280
#define BOFF_TE1 1536
#define BOFF_TE2 2048
#define BOFF_DE1 2304
#define BOFF_DE2 2816
#define BOFF_DE3 3072   /* 96, zero-padded past 95 */
#define BOFF_PB  3168   /* 2048 per-sample pred_bias */
#define BIAS_TOTAL 5216

// ---------------------------------------------------------------------------
// Persistent per-sample kernel: full 39-step recurrence, state in LDS.
// ---------------------------------------------------------------------------
__global__ __launch_bounds__(NT)
void etn_main(const float* __restrict__ past_root_vel, const float* __restrict__ past_quats,
              const float* __restrict__ past_root_offset, const float* __restrict__ past_quat_offset,
              const float* __restrict__ past_contacts,
              const float* __restrict__ target_root_pos, const float* __restrict__ target_quats,
              const float* __restrict__ init_root_pos, const float* __restrict__ pred_bias,
              const float* __restrict__ se_b1, const float* __restrict__ se_b2,
              const float* __restrict__ oe_b1, const float* __restrict__ oe_b2,
              const float* __restrict__ te_b1, const float* __restrict__ te_b2,
              const float* __restrict__ de_b1, const float* __restrict__ de_b2,
              const float* __restrict__ de_b3,
              const __bf16* __restrict__ predWt,
              const __bf16* __restrict__ se1, const __bf16* __restrict__ se2,
              const __bf16* __restrict__ oe1, const __bf16* __restrict__ oe2,
              const __bf16* __restrict__ te1, const __bf16* __restrict__ te2,
              const __bf16* __restrict__ dw1, const __bf16* __restrict__ dw2,
              const __bf16* __restrict__ dw3,
              float* __restrict__ pred_poses, float* __restrict__ pred_contacts) {
  const int s = blockIdx.x;
  const int tid = threadIdx.x;
  const int lane = tid & 31;
  const int wave = tid >> 5;

  __shared__ __align__(32) __bf16 sm_x[128];      // encoder input (padded)
  __shared__ __align__(32) __bf16 sm_hid[512];    // bf16 hidden vector
  __shared__ __align__(32) __bf16 sm_z[1280];     // LSTM input vector
  __shared__ __align__(16) float sm_bias[BIAS_TOTAL];
  __shared__ __align__(16) float sm_t512[512];
  __shared__ __align__(16) float sm_hstate[256], sm_hoffset[256], sm_htarget[256];
  __shared__ __align__(16) float sm_gates[2048];
  __shared__ __align__(16) float sm_h[512], sm_c[512];
  __shared__ __align__(16) float sm_d1[512], sm_d2[256], sm_out[96];
  __shared__ float in_quats[88], in_qoff[88], in_rvel[3], in_roff[3], in_cont[4];
  __shared__ float st_quats[88], st_rvel[3], st_cont[4];
  __shared__ float sm_glob[3];

  const __bf16* myW = predWt + (size_t)s * 128 * 40 * 512;

  // ---- stage all biases (zero-padded) into LDS once ----
  for (int i = tid; i < BIAS_TOTAL; i += NT) {
    float v = 0.f;
    if (i < BOFF_SE2)      v = se_b1[i];
    else if (i < BOFF_OE1) v = se_b2[i - BOFF_SE2];
    else if (i < BOFF_OE2) v = oe_b1[i - BOFF_OE1];
    else if (i < BOFF_TE1) v = oe_b2[i - BOFF_OE2];
    else if (i < BOFF_TE2) v = te_b1[i - BOFF_TE1];
    else if (i < BOFF_DE1) v = te_b2[i - BOFF_TE2];
    else if (i < BOFF_DE2) v = de_b1[i - BOFF_DE1];
    else if (i < BOFF_DE3) v = de_b2[i - BOFF_DE2];
    else if (i < BOFF_PB)  { int j = i - BOFF_DE3; v = (j < 95) ? de_b3[j] : 0.f; }
    else                   v = pred_bias[(size_t)s * 2048 + (i - BOFF_PB)];
    sm_bias[i] = v;
  }

  auto enc2 = [&](const __bf16* w1, int boff1, int nkt1,
                  const __bf16* w2, int boff2, float* out256) {
    wmma_mv<true>(w1, sm_x, sm_t512, sm_bias + boff1, 32, nkt1, lane, wave);
    __syncthreads();
    for (int i = tid; i < 512; i += NT) sm_hid[i] = (__bf16)sm_t512[i];
    __syncthreads();
    wmma_mv<true>(w2, sm_hid, out256, sm_bias + boff2, 16, 16, lane, wave);
    __syncthreads();
  };

  auto step = [&]() {
    // state_vec = [contacts(4), quats(88), rvel(3)] zero-padded to 96 (+pad 128)
    if (tid < 128) {
      float v = 0.f;
      if (tid < 4) v = in_cont[tid];
      else if (tid < 92) v = in_quats[tid - 4];
      else if (tid < 95) v = in_rvel[tid - 92];
      sm_x[tid] = (__bf16)v;
    }
    __syncthreads();
    enc2(se1, BOFF_SE1, 3, se2, BOFF_SE2, sm_hstate);

    // offset_vec = [roff(3), qoff(88)]
    if (tid < 128) {
      float v = 0.f;
      if (tid < 3) v = in_roff[tid];
      else if (tid < 91) v = in_qoff[tid - 3];
      sm_x[tid] = (__bf16)v;
    }
    __syncthreads();
    enc2(oe1, BOFF_OE1, 3, oe2, BOFF_OE2, sm_hoffset);

    // z = [h_state, h_offset, h_target, h]
    for (int i = tid; i < 1280; i += NT) {
      float v;
      if (i < 256) v = sm_hstate[i];
      else if (i < 512) v = sm_hoffset[i - 256];
      else if (i < 768) v = sm_htarget[i - 512];
      else v = sm_h[i - 768];
      sm_z[i] = (__bf16)v;
    }
    __syncthreads();

    // per-sample hypernetwork: gates = W_s @ z + bias (dominant, HBM-streamed)
    wmma_mv<false>(myW, sm_z, sm_gates, sm_bias + BOFF_PB, 128, 40, lane, wave);
    __syncthreads();

    // LSTM cell (torch gate order i,f,g,o)
    for (int i = tid; i < 512; i += NT) {
      float ig = sm_gates[i], fg = sm_gates[512 + i];
      float gg = sm_gates[1024 + i], og = sm_gates[1536 + i];
      float c2 = sigmf(fg) * sm_c[i] + sigmf(ig) * tanhf(gg);
      float h2 = sigmf(og) * tanhf(c2);
      sm_c[i] = c2; sm_h[i] = h2;
    }
    __syncthreads();

    // decoder: 512 -> 512 -> 256 -> 95(pad 96)
    for (int i = tid; i < 512; i += NT) sm_hid[i] = (__bf16)sm_h[i];
    __syncthreads();
    wmma_mv<true>(dw1, sm_hid, sm_d1, sm_bias + BOFF_DE1, 32, 16, lane, wave);
    __syncthreads();
    for (int i = tid; i < 512; i += NT) sm_hid[i] = (__bf16)sm_d1[i];
    __syncthreads();
    wmma_mv<true>(dw2, sm_hid, sm_d2, sm_bias + BOFF_DE2, 16, 16, lane, wave);
    __syncthreads();
    for (int i = tid; i < 512; i += NT) sm_hid[i] = (i < 256) ? (__bf16)sm_d2[i] : (__bf16)0.f;
    __syncthreads();
    wmma_mv<false>(dw3, sm_hid, sm_out, sm_bias + BOFF_DE3, 6, 8, lane, wave);
    __syncthreads();

    // residual update + contact sigmoid
    if (tid < 88) st_quats[tid] = sm_out[tid] + in_quats[tid];
    else if (tid < 91) st_rvel[tid - 88] = sm_out[tid] + in_rvel[tid - 88];
    else if (tid < 95) st_cont[tid - 91] = sigmf(sm_out[tid]);
    __syncthreads();
    // per-joint quat normalization (22 joints x 4)
    if (tid < 22) {
      float a = st_quats[4 * tid], b = st_quats[4 * tid + 1];
      float c = st_quats[4 * tid + 2], d = st_quats[4 * tid + 3];
      float inv = 1.f / fmaxf(sqrtf(a * a + b * b + c * c + d * d), 1e-12f);
      st_quats[4 * tid] = a * inv; st_quats[4 * tid + 1] = b * inv;
      st_quats[4 * tid + 2] = c * inv; st_quats[4 * tid + 3] = d * inv;
    }
    __syncthreads();
  };

  // ---- init ----
  for (int i = tid; i < 512; i += NT) { sm_h[i] = 0.f; sm_c[i] = 0.f; }
  if (tid < 3) sm_glob[tid] = init_root_pos[s * 3 + tid];
  if (tid < 128) {
    float v = (tid < 88) ? target_quats[(size_t)s * 88 + tid] : 0.f;
    sm_x[tid] = (__bf16)v;
  }
  __syncthreads();
  enc2(te1, BOFF_TE1, 3, te2, BOFF_TE2, sm_htarget);   // hoisted target encoding

  // ---- past phase: 10 steps, inputs from past arrays ----
  for (int t = 0; t < 10; ++t) {
    if (tid < 88) {
      in_quats[tid] = past_quats[((size_t)s * 10 + t) * 88 + tid];
      in_qoff[tid]  = past_quat_offset[((size_t)s * 10 + t) * 88 + tid];
    }
    if (tid < 3) {
      in_rvel[tid] = past_root_vel[((size_t)s * 10 + t) * 3 + tid];
      in_roff[tid] = past_root_offset[((size_t)s * 10 + t) * 3 + tid];
    }
    if (tid < 4) in_cont[tid] = past_contacts[((size_t)s * 10 + t) * 4 + tid];
    __syncthreads();
    step();
    if (tid < 3) sm_glob[tid] += in_rvel[tid];   // glob += input rvel
    __syncthreads();
  }

  // first frame = [nrv, nq]
  if (tid < 3)  pred_poses[(size_t)(s * 30) * 91 + tid] = st_rvel[tid];
  if (tid < 88) pred_poses[(size_t)(s * 30) * 91 + 3 + tid] = st_quats[tid];
  if (tid < 4)  pred_contacts[(size_t)(s * 30) * 4 + tid] = st_cont[tid];
  // carry into transition phase
  if (tid < 3) { in_rvel[tid] = st_rvel[tid]; in_roff[tid] = sm_glob[tid] - target_root_pos[s * 3 + tid]; }
  if (tid < 88) { in_quats[tid] = st_quats[tid]; in_qoff[tid] = st_quats[tid] - target_quats[(size_t)s * 88 + tid]; }
  if (tid < 4) in_cont[tid] = st_cont[tid];
  __syncthreads();

  // ---- transition phase: 29 steps ----
  for (int u = 0; u < 29; ++u) {
    step();
    if (tid < 3) sm_glob[tid] += st_rvel[tid];   // glob += NEW rvel
    __syncthreads();
    size_t fr = (size_t)(s * 30 + u + 1);
    if (tid < 3)  pred_poses[fr * 91 + tid] = sm_glob[tid];
    if (tid < 88) pred_poses[fr * 91 + 3 + tid] = st_quats[tid];
    if (tid < 4)  pred_contacts[fr * 4 + tid] = st_cont[tid];
    if (tid < 3) { in_rvel[tid] = st_rvel[tid]; in_roff[tid] = sm_glob[tid] - target_root_pos[s * 3 + tid]; }
    if (tid < 88) { in_quats[tid] = st_quats[tid]; in_qoff[tid] = st_quats[tid] - target_quats[(size_t)s * 88 + tid]; }
    if (tid < 4) in_cont[tid] = st_cont[tid];
    __syncthreads();
  }
}

// ---------------------------------------------------------------------------
extern "C" void kernel_launch(void* const* d_in, const int* in_sizes, int n_in,
                              void* d_out, int out_size, void* d_ws, size_t ws_size,
                              hipStream_t stream) {
  (void)in_sizes; (void)n_in; (void)out_size; (void)ws_size;
  const float* past_root_vel    = (const float*)d_in[0];
  const float* past_quats       = (const float*)d_in[1];
  const float* past_root_offset = (const float*)d_in[2];
  const float* past_quat_offset = (const float*)d_in[3];
  const float* past_contacts    = (const float*)d_in[4];
  const float* target_root_pos  = (const float*)d_in[5];
  const float* target_quats     = (const float*)d_in[6];
  const float* init_root_pos    = (const float*)d_in[7];
  const float* pred_weights     = (const float*)d_in[8];
  const float* pred_bias        = (const float*)d_in[9];
  const float* se_w1 = (const float*)d_in[10]; const float* se_b1 = (const float*)d_in[11];
  const float* se_w2 = (const float*)d_in[12]; const float* se_b2 = (const float*)d_in[13];
  const float* oe_w1 = (const float*)d_in[14]; const float* oe_b1 = (const float*)d_in[15];
  const float* oe_w2 = (const float*)d_in[16]; const float* oe_b2 = (const float*)d_in[17];
  const float* te_w1 = (const float*)d_in[18]; const float* te_b1 = (const float*)d_in[19];
  const float* te_w2 = (const float*)d_in[20]; const float* te_b2 = (const float*)d_in[21];
  const float* de_w1 = (const float*)d_in[22]; const float* de_b1 = (const float*)d_in[23];
  const float* de_w2 = (const float*)d_in[24]; const float* de_b2 = (const float*)d_in[25];
  const float* de_w3 = (const float*)d_in[26]; const float* de_b3 = (const float*)d_in[27];

  // workspace layout (bf16 elements)
  __bf16* ws = (__bf16*)d_ws;
  size_t off = 0;
  __bf16* predWt = ws + off; off += (size_t)128 * 128 * 40 * 512;   // 671 MB
  __bf16* se1 = ws + off; off += 32 * 3 * 512;
  __bf16* se2 = ws + off; off += 16 * 16 * 512;
  __bf16* oe1 = ws + off; off += 32 * 3 * 512;
  __bf16* oe2 = ws + off; off += 16 * 16 * 512;
  __bf16* te1 = ws + off; off += 32 * 3 * 512;
  __bf16* te2 = ws + off; off += 16 * 16 * 512;
  __bf16* dw1 = ws + off; off += 32 * 16 * 512;
  __bf16* dw2 = ws + off; off += 16 * 16 * 512;
  __bf16* dw3 = ws + off; off += 6 * 8 * 512;

  // per-launch bf16 re-tiling — halves steady-state HBM traffic
  {
    size_t total = (size_t)128 * 128 * 40 * 512;
    convert_pred<<<(unsigned)((total + NT - 1) / NT), NT, 0, stream>>>(pred_weights, predWt);
  }
  auto convW = [&](const float* W, __bf16* dst, int O, int K, int nrt, int nkt) {
    int total = nrt * nkt * 512;
    convert_wt<<<(total + NT - 1) / NT, NT, 0, stream>>>(W, dst, O, K, nrt, nkt);
  };
  convW(se_w1, se1, 512,  95, 32,  3);
  convW(se_w2, se2, 256, 512, 16, 16);
  convW(oe_w1, oe1, 512,  91, 32,  3);
  convW(oe_w2, oe2, 256, 512, 16, 16);
  convW(te_w1, te1, 512,  88, 32,  3);
  convW(te_w2, te2, 256, 512, 16, 16);
  convW(de_w1, dw1, 512, 512, 32, 16);
  convW(de_w2, dw2, 256, 512, 16, 16);
  convW(de_w3, dw3,  95, 256,  6,  8);

  float* pred_poses = (float*)d_out;                         // [128,30,91]
  float* pred_contacts = pred_poses + (size_t)128 * 30 * 91; // [128,30,4]

  etn_main<<<128, NT, 0, stream>>>(
      past_root_vel, past_quats, past_root_offset, past_quat_offset, past_contacts,
      target_root_pos, target_quats, init_root_pos, pred_bias,
      se_b1, se_b2, oe_b1, oe_b2, te_b1, te_b2, de_b1, de_b2, de_b3,
      predWt, se1, se2, oe1, oe2, te1, te2, dw1, dw2, dw3,
      pred_poses, pred_contacts);
}
